// MultiHeadSelfAttention_77309412028
// MI455X (gfx1250) — matrix operand order repaired and checked by reading, not compiled, not run on hardware
//
#include <hip/hip_runtime.h>
#include <hip/hip_bf16.h>

// MHA: B=4, S=2048, D=1024, H=16, HD=64
#define BB   4
#define SS   2048
#define DD   1024
#define HH   16
#define HDD  64

typedef _Float16 v16h __attribute__((ext_vector_type(16)));
typedef _Float16 half8 __attribute__((ext_vector_type(8)));
typedef float    v8f  __attribute__((ext_vector_type(8)));

static __device__ __forceinline__ v16h cat8(half8 lo, half8 hv) {
    v16h r;
#pragma unroll
    for (int e = 0; e < 8; ++e) { r[e] = lo[e]; r[e + 8] = hv[e]; }
    return r;
}

// ---------------------------------------------------------------------------
// Kernel 1: Y = X @ W^T + b  -> f16, head-major [B,H,S,HD]
// grid = (D/64, BS/128, 3); block = 256 (8 waves, 4x2, each wave 32x32 out)
// ---------------------------------------------------------------------------
#define AST 40   // LDS row stride in halves (80B, 16B multiple)
#define BST 40

__global__ __launch_bounds__(256)
void qkv_proj_kernel(const float* __restrict__ xq, const float* __restrict__ xk,
                     const float* __restrict__ xv,
                     const float* __restrict__ Wq, const float* __restrict__ bq,
                     const float* __restrict__ Wk, const float* __restrict__ bk,
                     const float* __restrict__ Wv, const float* __restrict__ bv,
                     _Float16* __restrict__ Qf, _Float16* __restrict__ Kf,
                     _Float16* __restrict__ Vf)
{
    __shared__ __align__(16) _Float16 As[128 * AST];
    __shared__ __align__(16) _Float16 Bs[64 * BST];

    const int z = blockIdx.z;
    const float* X    = (z == 0) ? xq : (z == 1) ? xk : xv;
    const float* W    = (z == 0) ? Wq : (z == 1) ? Wk : Wv;
    const float* bias = (z == 0) ? bq : (z == 1) ? bk : bv;
    _Float16*    dst  = (z == 0) ? Qf : (z == 1) ? Kf : Vf;

    const int Mtile = blockIdx.y * 128;
    const int Ntile = blockIdx.x * 64;
    const int tid  = threadIdx.x;
    const int lane = tid & 31;
    const int wv   = tid >> 5;
    const int l15  = lane & 15;
    const int hi   = lane >> 4;      // 0: lanes 0-15, 1: lanes 16-31
    const int wm   = wv >> 1, wn = wv & 1;

    v8f acc[2][2] = {};

    for (int kk = 0; kk < DD; kk += 32) {
        __syncthreads();
        // stage A: 128x32 fp32 -> f16 LDS
#pragma unroll
        for (int i = 0; i < 4; ++i) {
            int lin = tid + 256 * i;
            int row = lin >> 3;
            int c4  = (lin & 7) << 2;
            const float4 v = *(const float4*)(X + (size_t)(Mtile + row) * DD + kk + c4);
            _Float16* p = &As[row * AST + c4];
            p[0] = (_Float16)v.x; p[1] = (_Float16)v.y;
            p[2] = (_Float16)v.z; p[3] = (_Float16)v.w;
        }
        // stage B: 64x32 fp32 (rows of W = output cols) -> f16 LDS
#pragma unroll
        for (int i = 0; i < 2; ++i) {
            int lin = tid + 256 * i;
            int row = lin >> 3;
            int c4  = (lin & 7) << 2;
            const float4 v = *(const float4*)(W + (size_t)(Ntile + row) * DD + kk + c4);
            _Float16* p = &Bs[row * BST + c4];
            p[0] = (_Float16)v.x; p[1] = (_Float16)v.y;
            p[2] = (_Float16)v.z; p[3] = (_Float16)v.w;
        }
        __syncthreads();

        v16h a[2], b[2];
#pragma unroll
        for (int i = 0; i < 2; ++i) {   // A 16x32: lane<16 K{0..7,16..23}, lane>=16 K{8..15,24..31}
            int row = 32 * wm + 16 * i + l15;
            half8 lo = *(const half8*)&As[row * AST + hi * 8];
            half8 hv = *(const half8*)&As[row * AST + 16 + hi * 8];
            a[i] = cat8(lo, hv);
        }
#pragma unroll
        for (int j = 0; j < 2; ++j) {   // B 32x16: lane<16 K{0..7,16..23}, lane>=16 K{8..15,24..31}
            int row = 32 * wn + 16 * j + l15;
            half8 lo = *(const half8*)&Bs[row * BST + hi * 8];
            half8 hv = *(const half8*)&Bs[row * BST + 16 + hi * 8];
            b[j] = cat8(lo, hv);
        }
#pragma unroll
        for (int i = 0; i < 2; ++i)
#pragma unroll
            for (int j = 0; j < 2; ++j)
                acc[i][j] = __builtin_amdgcn_wmma_f32_16x16x32_f16(
                    false, a[i], false, b[j], (short)0, acc[i][j], false, false);
    }

    // epilogue: + bias, convert f16, scatter to [B,H,S,HD]
#pragma unroll
    for (int i = 0; i < 2; ++i) {
#pragma unroll
        for (int j = 0; j < 2; ++j) {
#pragma unroll
            for (int r = 0; r < 8; ++r) {
                int m = Mtile + 32 * wm + 16 * i + r + 8 * hi;   // C layout: VGPR r -> M=r / r+8
                int n = Ntile + 32 * wn + 16 * j + l15;
                float v = acc[i][j][r] + bias[n];
                int bb = m >> 11, s = m & (SS - 1);
                int hh = n >> 6,  hd = n & (HDD - 1);
                dst[(size_t)(((bb << 4) + hh) * SS + s) * HDD + hd] = (_Float16)v;
            }
        }
    }
}

// ---------------------------------------------------------------------------
// Kernel 2: flash attention per (b,h). grid = (S/128, B*H); block = 256.
// Each wave owns 16 query rows; loop over 32-key tiles with online softmax.
// ---------------------------------------------------------------------------
#define KST 80   // K tile LDS row stride (halves)
#define VST 40   // V^T tile LDS row stride
#define PST 40   // per-wave P scratch row stride

__global__ __launch_bounds__(256)
void attn_kernel(const _Float16* __restrict__ Qf, const _Float16* __restrict__ Kf,
                 const _Float16* __restrict__ Vf, float* __restrict__ out)
{
    __shared__ __align__(16) _Float16 Ks[32 * KST];       // [key][hd]
    __shared__ __align__(16) _Float16 Vt[64 * VST];       // [hd][key]
    __shared__ __align__(16) _Float16 Pb[8 * 16 * PST];   // per-wave P (16x32)

    const int bh  = blockIdx.y;
    const int b   = bh >> 4, h = bh & 15;
    const int tid = threadIdx.x;
    const int lane = tid & 31, wv = tid >> 5;
    const int l15 = lane & 15, hi = lane >> 4;
    const int q0  = blockIdx.x * 128 + 16 * wv;

    const _Float16* Qb = Qf + (size_t)bh * SS * HDD;
    const _Float16* Kb = Kf + (size_t)bh * SS * HDD;
    const _Float16* Vb = Vf + (size_t)bh * SS * HDD;

    // Q A-fragments (16 rows x 64 hd = two 16x32 frags), loaded once
    v16h aq[2];
#pragma unroll
    for (int k = 0; k < 2; ++k) {
        const _Float16* qr = Qb + (size_t)(q0 + l15) * HDD + 32 * k;
        half8 lo = *(const half8*)(qr + hi * 8);
        half8 hv = *(const half8*)(qr + 16 + hi * 8);
        aq[k] = cat8(lo, hv);
    }

    v8f o[4] = {};
    float m_run[8], l_run[8];
#pragma unroll
    for (int r = 0; r < 8; ++r) { m_run[r] = -3.0e38f; l_run[r] = 0.0f; }

    for (int kt = 0; kt < SS; kt += 32) {
        if (kt + 32 < SS) {  // prefetch next tile (global_prefetch_b8)
            __builtin_prefetch(Kb + (size_t)(kt + 32 + (tid >> 3)) * HDD, 0, 1);
            __builtin_prefetch(Vb + (size_t)(kt + 32 + (tid >> 3)) * HDD, 0, 1);
        }
        __syncthreads();
        {   // stage K row-major, V transposed
            int key = tid >> 3;
            int c8  = (tid & 7) << 3;
            half8 kv = *(const half8*)(Kb + (size_t)(kt + key) * HDD + c8);
            *(half8*)&Ks[key * KST + c8] = kv;
            half8 vw = *(const half8*)(Vb + (size_t)(kt + key) * HDD + c8);
#pragma unroll
            for (int e = 0; e < 8; ++e) Vt[(c8 + e) * VST + key] = vw[e];
        }
        __syncthreads();

        // scores: 16 queries x 32 keys = two 16x16 C tiles, contraction hd=64
        v8f c[2];
#pragma unroll
        for (int n = 0; n < 2; ++n) {
            v8f zz = {};
#pragma unroll
            for (int k = 0; k < 2; ++k) {
                const _Float16* kr = &Ks[(16 * n + l15) * KST + 32 * k + hi * 8];
                v16h bf = cat8(*(const half8*)kr, *(const half8*)(kr + 16));
                zz = __builtin_amdgcn_wmma_f32_16x16x32_f16(
                    false, aq[k], false, bf, (short)0, zz, false, false);
            }
            c[n] = zz;
        }

        // online softmax (row r lives in VGPR r across one 16-lane half)
#pragma unroll
        for (int r = 0; r < 8; ++r) {
            float s0 = c[0][r] * 0.125f, s1 = c[1][r] * 0.125f;  // 1/sqrt(64)
            float mx = fmaxf(s0, s1);
#pragma unroll
            for (int msk = 1; msk < 16; msk <<= 1)
                mx = fmaxf(mx, __shfl_xor(mx, msk, 32));
            float newm = fmaxf(m_run[r], mx);
            float p0 = __expf(s0 - newm), p1 = __expf(s1 - newm);
            float sum = p0 + p1;
#pragma unroll
            for (int msk = 1; msk < 16; msk <<= 1)
                sum += __shfl_xor(sum, msk, 32);
            float fac = __expf(m_run[r] - newm);
            m_run[r] = newm;
            l_run[r] = l_run[r] * fac + sum;
#pragma unroll
            for (int j = 0; j < 4; ++j) o[j][r] *= fac;
            c[0][r] = p0; c[1][r] = p1;
        }

        // C-layout -> A-layout via per-wave LDS scratch
        _Float16* pb = &Pb[wv * 16 * PST];
#pragma unroll
        for (int n = 0; n < 2; ++n)
#pragma unroll
            for (int r = 0; r < 8; ++r)
                pb[(r + 8 * hi) * PST + 16 * n + l15] = (_Float16)c[n][r];

        {
            const _Float16* pr = pb + l15 * PST;
            v16h ap = cat8(*(const half8*)(pr + hi * 8),
                           *(const half8*)(pr + 16 + hi * 8));
#pragma unroll
            for (int j = 0; j < 4; ++j) {  // o[16q x 64hd] += P(16x32) x V(32 x 16hd)
                const _Float16* vr = &Vt[(16 * j + l15) * VST + hi * 8];
                v16h bf = cat8(*(const half8*)vr, *(const half8*)(vr + 16));
                o[j] = __builtin_amdgcn_wmma_f32_16x16x32_f16(
                    false, ap, false, bf, (short)0, o[j], false, false);
            }
        }
    }

    // normalize and store fp32 [B,S,D]
#pragma unroll
    for (int r = 0; r < 8; ++r) {
        float inv = 1.0f / l_run[r];
        int sq = q0 + r + 8 * hi;
#pragma unroll
        for (int j = 0; j < 4; ++j) {
            int d = h * HDD + 16 * j + l15;
            out[(size_t)(b * SS + sq) * DD + d] = o[j][r] * inv;
        }
    }
}

// ---------------------------------------------------------------------------
extern "C" void kernel_launch(void* const* d_in, const int* in_sizes, int n_in,
                              void* d_out, int out_size, void* d_ws, size_t ws_size,
                              hipStream_t stream) {
    const float* query = (const float*)d_in[0];
    const float* key   = (const float*)d_in[1];
    const float* value = (const float*)d_in[2];
    const float* Wq    = (const float*)d_in[3];
    const float* bq    = (const float*)d_in[4];
    const float* Wk    = (const float*)d_in[5];
    const float* bk    = (const float*)d_in[6];
    const float* Wv    = (const float*)d_in[7];
    const float* bv    = (const float*)d_in[8];

    const size_t per = (size_t)BB * HH * SS * HDD;   // 8,388,608 halves = 16 MiB
    _Float16* Qf = (_Float16*)d_ws;
    _Float16* Kf = Qf + per;
    _Float16* Vf = Kf + per;                         // total ws use: 48 MiB

    dim3 gp(DD / 64, (BB * SS) / 128, 3);            // (16, 64, 3)
    qkv_proj_kernel<<<gp, 256, 0, stream>>>(query, key, value,
                                            Wq, bq, Wk, bk, Wv, bv,
                                            Qf, Kf, Vf);

    dim3 ga(SS / 128, BB * HH);                      // (16, 64)
    attn_kernel<<<ga, 256, 0, stream>>>(Qf, Kf, Vf, (float*)d_out);
}
